// WindowAttention_32624571581017
// MI455X (gfx1250) — matrix operand order repaired and checked by reading, not compiled, not run on hardware
//
#include <hip/hip_runtime.h>

// Fused Swin window-attention forward for gfx1250 (MI455X).
// One workgroup (256 thr = 8 wave32) per window. f16 WMMA, f32 accumulate.
// - All WMMA fragments are contiguous ds_load_b128 pairs.
// - Weights are pre-converted once (prep kernel) to transposed/padded f16 in
//   d_ws, then staged into LDS with GLOBAL_LOAD_ASYNC_TO_LDS_B128 (ASYNCcnt).

#define DIMC 192
#define HEADS 6
#define HD 32
#define TT 49
#define BNW 4096
#define NW 64
#define TP 64

#define XS  200   // stride (halfs) for Xs/Qs/Ks/Os: mult of 8, 16B rows
#define WTS 40    // transposed-weight stride (32 k + 8 pad)
#define SSS 65    // f32 scores stride
#define PSS 72    // f16 probs stride (mult of 8)
#define VTS 72    // Vt stride (64 tokens + pad, mult of 8)

// LDS layout (bytes, all 16B aligned):
#define OFF_X   0        // Xs 64x200 f16 (25600)      | later Os
#define OFF_W   25600    // Wt 576x40 f16 (46080)      | Ss+Ps | Wpt 192x40
#define OFF_SS  25600    // Ss 64x65 f32 (16640)
#define OFF_PS  42240    // Ps 64x72 f16 (9216)
#define OFF_Q   71680    // Qs 64x200 f16 (25600)
#define OFF_K   97280    // Ks 64x200 f16 (25600)
#define OFF_VT  122880   // Vt 192x72 f16 (27648)
#define OFF_TM  150528   // tm 64 f32
#define OFF_BQ  150784   // bq 576 f32
#define OFF_BP  153088   // bp 192 f32
#define SMEM_SZ 153856

// d_ws layout: pre-transposed padded f16 weights
#define WT_CHUNK_H   (576 * WTS)            // halfs per qkv k-chunk
#define WPT_CHUNK_H  (192 * WTS)            // halfs per proj k-chunk
#define WT_BYTES     (6 * WT_CHUNK_H * 2)   // 276480
#define WPT_BYTES    (6 * WPT_CHUNK_H * 2)  // 92160
#define WS_NEEDED    (WT_BYTES + WPT_BYTES) // 368640

typedef __attribute__((ext_vector_type(4)))  _Float16 v4h;
typedef __attribute__((ext_vector_type(8)))  _Float16 v8h;
typedef __attribute__((ext_vector_type(16))) _Float16 v16h;
typedef __attribute__((ext_vector_type(8)))  float    v8f;

__device__ __forceinline__ v16h cat16(v8h lo, v8h hi) {
  return __builtin_shufflevector(lo, hi, 0, 1, 2, 3, 4, 5, 6, 7,
                                         8, 9, 10, 11, 12, 13, 14, 15);
}

__device__ __forceinline__ v8f wmma16(v16h a, v16h b, v8f c) {
  return __builtin_amdgcn_wmma_f32_16x16x32_f16(false, a, false, b, (short)0, c,
                                                false, false);
}

// A 16x32 f16 fragment from row-major M (stride mult of 8 halfs, 16B base).
__device__ __forceinline__ v16h load_a_vec(const _Float16* base, int stride, int lane) {
  const int m = lane & 15, hf = (lane >> 4) & 1;
  const _Float16* r = base + m * stride + 8 * hf;
  v8h lo = *(const v8h*)(r);
  v8h hi = *(const v8h*)(r + 16);
  return cat16(lo, hi);
}

// B 32x16 f16 fragment where B = M^T (M row-major): k = 8hf+0..7, 16+8hf+0..7.
__device__ __forceinline__ v16h load_bt_vec(const _Float16* base, int stride, int lane) {
  const int n = lane & 15, hf = (lane >> 4) & 1;
  const _Float16* p = base + n * stride + 8 * hf;
  v8h lo = *(const v8h*)(p);
  v8h hi = *(const v8h*)(p + 16);
  return cat16(lo, hi);
}

// 16B global -> LDS async copy (GV mode, tracked by ASYNCcnt).
// LDS byte offset = low 32 bits of the flat shared address (ISA 10.2).
__device__ __forceinline__ void async_copy_b128(unsigned lds_off, const void* gsrc) {
  asm volatile("global_load_async_to_lds_b128 %0, %1, off"
               :: "v"(lds_off), "v"(gsrc) : "memory");
}
__device__ __forceinline__ void wait_async0() {
  asm volatile("s_wait_asynccnt 0x0" ::: "memory");
}

// ---- prep: w_qkv/w_proj f32 -> transposed padded f16 chunks in d_ws ----
__global__ void __launch_bounds__(256)
prep_weights(const float* __restrict__ wqkv, const float* __restrict__ wproj,
             _Float16* __restrict__ wt, _Float16* __restrict__ wpt) {
  int i = blockIdx.x * 256 + threadIdx.x;
  if (i < 6 * 32 * 576) {                       // wt[kc][c*40 + kr]
    int kc = i / 18432, r = i % 18432, kr = r / 576, c = r % 576;
    wt[kc * WT_CHUNK_H + c * WTS + kr] = (_Float16)wqkv[i];
  } else {
    int j = i - 6 * 32 * 576;
    if (j < 6 * 32 * 192) {                     // wpt[kc][c*40 + kr]
      int kc = j / 6144, r = j % 6144, kr = r / 192, c = r % 192;
      wpt[kc * WPT_CHUNK_H + c * WTS + kr] = (_Float16)wproj[j];
    }
  }
}

template <bool FASTW>
__global__ void __launch_bounds__(256)
win_attn_fused(const float* __restrict__ x,
               const float* __restrict__ token_mask,
               const float* __restrict__ rel_pos_bias,
               const float* __restrict__ shift_attn_mask,
               const float* __restrict__ w_qkv,
               const float* __restrict__ b_qkv,
               const float* __restrict__ w_proj,
               const float* __restrict__ b_proj,
               const _Float16* __restrict__ wt_g,
               const _Float16* __restrict__ wpt_g,
               float* __restrict__ out)
{
  __shared__ __align__(16) char smem[SMEM_SZ];
  _Float16* Xs  = (_Float16*)(smem + OFF_X);
  _Float16* Os  = (_Float16*)(smem + OFF_X);    // alias: Xs dead after GEMM1
  _Float16* Wt  = (_Float16*)(smem + OFF_W);
  float*    Ss  = (float*)   (smem + OFF_SS);   // alias with Wt region
  _Float16* Ps  = (_Float16*)(smem + OFF_PS);
  _Float16* Wpt = (_Float16*)(smem + OFF_W);
  _Float16* Qs  = (_Float16*)(smem + OFF_Q);
  _Float16* Ks  = (_Float16*)(smem + OFF_K);
  _Float16* Vt  = (_Float16*)(smem + OFF_VT);
  float*    tm  = (float*)   (smem + OFF_TM);
  float*    bq  = (float*)   (smem + OFF_BQ);
  float*    bp  = (float*)   (smem + OFF_BP);

  const int b    = blockIdx.x;
  const int tid  = threadIdx.x;
  const int lane = tid & 31;
  const int wave = tid >> 5;
  const int l16  = lane & 15;
  const int hf   = (lane >> 4) & 1;
  const int wi   = b & (NW - 1);
  const float scale = 0.17677669529663687f;     // 1/sqrt(32)
  const v8f vzero = {0.f, 0.f, 0.f, 0.f, 0.f, 0.f, 0.f, 0.f};
  const unsigned wlds = (unsigned)(uintptr_t)(smem + OFF_W);

  // ---- stage x tile (float4 -> 4 halfs), zero pad rows, biases, mask ----
  {
    const v8h z8 = {};
    for (int idx = tid; idx < (TP - TT) * XS / 8; idx += 256)
      *(v8h*)(Xs + TT * XS + idx * 8) = z8;
    const float4* x4 = (const float4*)(x + (size_t)b * (TT * DIMC));
    for (int idx = tid; idx < TT * DIMC / 4; idx += 256) {
      int r = idx / 48, c4 = idx % 48;
      float4 v = x4[idx];
      v4h hv = {(_Float16)v.x, (_Float16)v.y, (_Float16)v.z, (_Float16)v.w};
      *(v4h*)(Xs + r * XS + c4 * 4) = hv;
    }
  }
  for (int idx = tid; idx < 3 * DIMC; idx += 256) bq[idx] = b_qkv[idx];
  if (tid < DIMC) bp[tid] = b_proj[tid];
  if (tid < TP)   tm[tid] = (tid < TT) ? token_mask[b * TT + tid] : 0.f;

  // ---- GEMM1 (transposed): QKV^T(576x64) = Wqkv^T(576x192) @ X^T(192x64) ----
  const int mt1 = wave >> 1;
  const int ntb = (wave & 1) * 18;
  v8f acc[18];
#pragma unroll
  for (int i = 0; i < 18; ++i) acc[i] = vzero;
  for (int kc = 0; kc < 6; ++kc) {
    if constexpr (FASTW) {
      const char* src = (const char*)(wt_g + kc * WT_CHUNK_H);
      for (int idx = tid; idx < WT_CHUNK_H / 8; idx += 256)
        async_copy_b128(wlds + idx * 16, src + idx * 16);
      wait_async0();
    } else {
      for (int idx = tid; idx < 32 * 576; idx += 256) {
        int kr = idx / 576, c = idx % 576;
        Wt[c * WTS + kr] = (_Float16)w_qkv[kc * (32 * 576) + idx];
      }
    }
    __syncthreads();
    v16h bfrag = load_bt_vec(Xs + mt1 * 16 * XS + kc * 32, XS, lane);
    v16h a_cur = load_a_vec(Wt + ntb * 16 * WTS, WTS, lane);
#pragma unroll
    for (int i = 0; i < 18; ++i) {
      v16h a_nxt = a_cur;
      if (i < 17) a_nxt = load_a_vec(Wt + (ntb + i + 1) * 16 * WTS, WTS, lane);
      acc[i] = wmma16(a_cur, bfrag, acc[i]);
      a_cur = a_nxt;
    }
    __syncthreads();
  }
  // epilogue: per lane token fixed (t = mt*16+l16), channels vary with vgpr.
  {
    const int t_tok = mt1 * 16 + l16;
    const float tsc = tm[t_tok] * scale;
#pragma unroll
    for (int i = 0; i < 18; ++i) {
      int nt = ntb + i;
      int sec = nt / 12;                        // 12 tiles per 192-ch section
#pragma unroll
      for (int v = 0; v < 8; ++v) {
        int n = nt * 16 + v + 8 * hf;
        int ch = n - sec * DIMC;
        float val = acc[i][v] + bq[n];
        if (sec == 0)      Qs[t_tok * XS + ch] = (_Float16)(val * tsc);
        else if (sec == 1) Ks[t_tok * XS + ch] = (_Float16)val;
        else               Vt[ch * VTS + t_tok] = (_Float16)val;
      }
    }
  }
  __syncthreads();

  // ---- attention, one head at a time ----
  for (int h = 0; h < HEADS; ++h) {
    // scores: S[t][s] via A=K(s,d) x B=Q^T(d,t); hd=32 -> 1 wmma per tile
    {
      const int tt = wave >> 1;
      const int t_ = tt * 16 + l16;
      v16h bqf = load_bt_vec(Qs + tt * 16 * XS + h * HD, XS, lane);
      const int st0 = (wave & 1) * 2;
      v16h a0 = load_a_vec(Ks + st0 * 16 * XS + h * HD, XS, lane);
      v16h a1 = load_a_vec(Ks + (st0 + 1) * 16 * XS + h * HD, XS, lane);
      v8f s0 = wmma16(a0, bqf, vzero);
      v8f s1 = wmma16(a1, bqf, vzero);
      const float* rpb = rel_pos_bias + (h * TT + t_) * TT;
      const float* sam = shift_attn_mask + (wi * TT + t_) * TT;
#pragma unroll
      for (int i = 0; i < 2; ++i) {
        v8f sv = i ? s1 : s0;
        int st = st0 + i;
#pragma unroll
        for (int v = 0; v < 8; ++v) {
          int s_ = st * 16 + v + 8 * hf;
          float val = -1e30f;                   // mask padded rows/cols
          if (t_ < TT && s_ < TT) val = sv[v] + rpb[s_] + sam[s_];
          Ss[t_ * SSS + s_] = val;
        }
      }
    }
    __syncthreads();
    // softmax over s: one thread per row (waves 0-1 fully active, uniform)
    if (tid < TP) {
      float mx = -3.4e38f;
      for (int s2 = 0; s2 < TP; ++s2) mx = fmaxf(mx, Ss[tid * SSS + s2]);
      float sum = 0.f;
      for (int s2 = 0; s2 < TP; ++s2) {
        float e = __expf(Ss[tid * SSS + s2] - mx);
        Ss[tid * SSS + s2] = e;
        sum += e;
      }
      float rs = 1.f / sum;
      for (int s2 = 0; s2 < TP; ++s2)
        Ps[tid * PSS + s2] = (_Float16)(Ss[tid * SSS + s2] * rs);
    }
    __syncthreads();
    // O_h(64x32) = P(64x64) @ V_h(64x32): A=P row-major, B=Vt^T; 1 tile/wave
    {
      int mt = wave >> 1, dt = wave & 1;
      v16h a0 = load_a_vec(Ps + mt * 16 * PSS, PSS, lane);
      v16h b0 = load_bt_vec(Vt + (h * HD + dt * 16) * VTS, VTS, lane);
      v16h a1 = load_a_vec(Ps + mt * 16 * PSS + 32, PSS, lane);
      v16h b1 = load_bt_vec(Vt + (h * HD + dt * 16) * VTS + 32, VTS, lane);
      v8f o = wmma16(a0, b0, vzero);
      o = wmma16(a1, b1, o);
      int dcol = h * HD + dt * 16 + l16;
#pragma unroll
      for (int v = 0; v < 8; ++v)
        Os[(mt * 16 + v + 8 * hf) * XS + dcol] = (_Float16)o[v];
    }
    __syncthreads();
  }

  // ---- proj (transposed): Out^T(192x64) = Wp^T(192x192) @ O^T(192x64) ----
  const int mtp = wave >> 1;
  const int npb = (wave & 1) * 6;
  v8f pacc[6];
#pragma unroll
  for (int i = 0; i < 6; ++i) pacc[i] = vzero;
  for (int kc = 0; kc < 6; ++kc) {
    if constexpr (FASTW) {
      const char* src = (const char*)(wpt_g + kc * WPT_CHUNK_H);
      for (int idx = tid; idx < WPT_CHUNK_H / 8; idx += 256)
        async_copy_b128(wlds + idx * 16, src + idx * 16);
      wait_async0();
    } else {
      for (int idx = tid; idx < 32 * DIMC; idx += 256) {
        int kr = idx / DIMC, c = idx % DIMC;
        Wpt[c * WTS + kr] = (_Float16)w_proj[kc * (32 * DIMC) + idx];
      }
    }
    __syncthreads();
    v16h bfrag = load_bt_vec(Os + mtp * 16 * XS + kc * 32, XS, lane);
    v16h a_cur = load_a_vec(Wpt + npb * 16 * WTS, WTS, lane);
#pragma unroll
    for (int i = 0; i < 6; ++i) {
      v16h a_nxt = a_cur;
      if (i < 5) a_nxt = load_a_vec(Wpt + (npb + i + 1) * 16 * WTS, WTS, lane);
      pacc[i] = wmma16(a_cur, bfrag, pacc[i]);
      a_cur = a_nxt;
    }
    __syncthreads();
  }
  // epilogue: per lane token fixed; 8 consecutive out channels per tile
  {
    const int t_ = mtp * 16 + l16;
    if (t_ < TT) {
      float* op = out + (size_t)b * (TT * DIMC) + t_ * DIMC;
#pragma unroll
      for (int i = 0; i < 6; ++i) {
        int nt = npb + i;
#pragma unroll
        for (int v = 0; v < 8; ++v) {
          int n = nt * 16 + v + 8 * hf;
          op[n] = pacc[i][v] + bp[n];
        }
      }
    }
  }
}

extern "C" void kernel_launch(void* const* d_in, const int* in_sizes, int n_in,
                              void* d_out, int out_size, void* d_ws, size_t ws_size,
                              hipStream_t stream) {
  (void)in_sizes; (void)n_in; (void)out_size;
  const float* xx  = (const float*)d_in[0];
  const float* tmk = (const float*)d_in[1];
  const float* rpb = (const float*)d_in[2];
  const float* sam = (const float*)d_in[3];
  const float* wq  = (const float*)d_in[4];
  const float* bqv = (const float*)d_in[5];
  const float* wp  = (const float*)d_in[6];
  const float* bpv = (const float*)d_in[7];
  float* o = (float*)d_out;
  if (ws_size >= (size_t)WS_NEEDED) {
    _Float16* wt  = (_Float16*)d_ws;
    _Float16* wpt = (_Float16*)((char*)d_ws + WT_BYTES);
    prep_weights<<<576, 256, 0, stream>>>(wq, wp, wt, wpt);
    win_attn_fused<true><<<BNW, 256, 0, stream>>>(
        xx, tmk, rpb, sam, wq, bqv, wp, bpv, wt, wpt, o);
  } else {
    win_attn_fused<false><<<BNW, 256, 0, stream>>>(
        xx, tmk, rpb, sam, wq, bqv, wp, bpv, nullptr, nullptr, o);
  }
}
